// QuantizedLinear_19490561589707
// MI455X (gfx1250) — 8-bit matrix operand order repaired and checked by reading, compiled, not run on hardware
//
#include <hip/hip_runtime.h>
#include <stdint.h>

typedef int v8i  __attribute__((ext_vector_type(8)));
typedef int v4i_ __attribute__((ext_vector_type(4)));
typedef int v2i_ __attribute__((ext_vector_type(2)));

#define B_DIM   8192
#define IN_DIM  4096
#define OUT_DIM 4096
#define KSTEP   64

__device__ __forceinline__ int imaxd(int a, int b) { return a > b ? a : b; }
__device__ __forceinline__ int iabsd(int a) { return a < 0 ? -a : a; }

__global__ void pack_x_kernel(const float* __restrict__ xq, uint32_t* __restrict__ xp) {
    size_t d = (size_t)blockIdx.x * blockDim.x + threadIdx.x;
    float4 f = ((const float4*)xq)[d];
    uint32_t p = ((uint32_t)(int)f.x & 0xFFu)
               | (((uint32_t)(int)f.y & 0xFFu) << 8)
               | (((uint32_t)(int)f.z & 0xFFu) << 16)
               | (((uint32_t)(int)f.w & 0xFFu) << 24);
    xp[d] = p;
}

__global__ void pack_w_kernel(const float* __restrict__ wq, int8_t* __restrict__ wp,
                              int* __restrict__ rowsum) {
    __shared__ int sred[256];
    const int o = blockIdx.x;
    const int t = threadIdx.x;
    const float* row = wq + (size_t)o * IN_DIM;
    const int base = t * 16;
    int acc = 0;
    uint32_t d[4];
#pragma unroll
    for (int g = 0; g < 4; ++g) {
        float4 f = *(const float4*)(row + base + g * 4);
        int a0 = (int)f.x, a1 = (int)f.y, a2 = (int)f.z, a3 = (int)f.w;
        acc += a0 + a1 + a2 + a3;
        d[g] = ((uint32_t)a0 & 0xFFu) | (((uint32_t)a1 & 0xFFu) << 8)
             | (((uint32_t)a2 & 0xFFu) << 16) | (((uint32_t)a3 & 0xFFu) << 24);
    }
    *(uint4*)(wp + (size_t)o * IN_DIM + base) = make_uint4(d[0], d[1], d[2], d[3]);

    sred[t] = acc;
    __syncthreads();
    for (int s = 128; s > 0; s >>= 1) {
        if (t < s) sred[t] += sred[t + s];
        __syncthreads();
    }
    if (t == 0) rowsum[o] = sred[0];
}

__global__ void zero_smax_kernel(int* smax) { *smax = 0; }

__global__ void __launch_bounds__(256)
gemm_kernel(const uint8_t* __restrict__ xp, const int8_t* __restrict__ wp,
            const int* __restrict__ rowsum, const float* __restrict__ x_scale,
            const float* __restrict__ x_zp, const float* __restrict__ fp_bias,
            const float* __restrict__ w_scale, int* __restrict__ lin_out,
            int* __restrict__ smax) {
    const int tid   = threadIdx.x;
    const int lane  = tid & 31;
    const int wave  = tid >> 5;
    const int wm    = wave & 1;
    const int wn    = wave >> 1;
    const int m_base = blockIdx.x * 128 + wm * 64;
    const int n_base = blockIdx.y * 128 + wn * 32;
    const int lrow   = lane & 15;
    const int lhalf  = lane >> 4;

    v8i C[4][2];
#pragma unroll
    for (int i = 0; i < 4; ++i)
#pragma unroll
        for (int j = 0; j < 2; ++j)
#pragma unroll
            for (int e = 0; e < 8; ++e) C[i][j][e] = 0;

    const uint8_t* aBase[4];
#pragma unroll
    for (int i = 0; i < 4; ++i)
        aBase[i] = xp + (size_t)(m_base + i * 16 + lrow) * IN_DIM + (lhalf << 4);
    const int8_t* bBase[2];
#pragma unroll
    for (int j = 0; j < 2; ++j)
        bBase[j] = wp + (size_t)(n_base + j * 16 + lrow) * IN_DIM + (lhalf << 4);

#pragma unroll 1
    for (int k0 = 0; k0 < IN_DIM; k0 += KSTEP) {
        v8i A[4];
#pragma unroll
        for (int i = 0; i < 4; ++i) {
            const uint8_t* p = aBase[i] + k0;
            v2i_ p0 = *(const v2i_*)(p);
            v2i_ p1 = *(const v2i_*)(p + 8);
            v2i_ p2 = *(const v2i_*)(p + 32);
            v2i_ p3 = *(const v2i_*)(p + 40);
            v8i a;
            a[0] = p0[0]; a[1] = p0[1]; a[2] = p1[0]; a[3] = p1[1];
            a[4] = p2[0]; a[5] = p2[1]; a[6] = p3[0]; a[7] = p3[1];
            A[i] = a;
        }
        v8i Bf[2];
#pragma unroll
        for (int j = 0; j < 2; ++j) {
            const int8_t* p = bBase[j] + k0;
            v4i_ q0 = *(const v4i_*)(p);
            v4i_ q1 = *(const v4i_*)(p + 32);
            v8i b;
            b[0] = q0[0]; b[1] = q0[1]; b[2] = q0[2]; b[3] = q0[3];
            b[4] = q1[0]; b[5] = q1[1]; b[6] = q1[2]; b[7] = q1[3];
            Bf[j] = b;
        }
#pragma unroll
        for (int i = 0; i < 4; ++i)
#pragma unroll
            for (int j = 0; j < 2; ++j)
                C[i][j] = __builtin_amdgcn_wmma_i32_16x16x64_iu8(
                     false, A[i],  true, Bf[j], C[i][j],
                     false,  false);
    }

    const float ws = w_scale[0];
    int n_of[2], rs[2];
    float fb[2];
#pragma unroll
    for (int j = 0; j < 2; ++j) {
        n_of[j] = n_base + j * 16 + lrow;
        rs[j]   = rowsum[n_of[j]];
        fb[j]   = fp_bias[n_of[j]];
    }
    int amax = 0;
#pragma unroll
    for (int i = 0; i < 4; ++i) {
#pragma unroll
        for (int v = 0; v < 8; ++v) {
            const int m   = m_base + i * 16 + (lhalf << 3) + v;
            const int zp  = (int)x_zp[m];
            const float inv = 1.0f / (ws * x_scale[m]);
#pragma unroll
            for (int j = 0; j < 2; ++j) {
                int lin = C[i][j][v] - zp * rs[j] + (int)rintf(fb[j] * inv);
                lin_out[(size_t)m * OUT_DIM + n_of[j]] = lin;
                amax = imaxd(amax, iabsd(lin));
            }
        }
    }

    __shared__ int sred[256];
    sred[tid] = amax;
    __syncthreads();
    for (int s = 128; s > 0; s >>= 1) {
        if (tid < s) sred[tid] = imaxd(sred[tid], sred[tid + s]);
        __syncthreads();
    }
    if (tid == 0) atomicMax(smax, sred[0]);
}

__global__ void requant_kernel(int* __restrict__ lin, const float* __restrict__ x_scale,
                               const float* __restrict__ w_scale,
                               const int* __restrict__ smax, float* __restrict__ out) {
    size_t gid = (size_t)blockIdx.x * blockDim.x + threadIdx.x;
    const float s = fmaxf((float)(*smax) / 127.0f, 1e-30f);
    if (gid < (size_t)B_DIM * OUT_DIM) {
        int l = lin[gid];
        float q = rintf((float)l / s);
        q = fminf(fmaxf(q, -127.0f), 127.0f);
        out[gid] = q;
    }
    if (gid < B_DIM) {
        out[(size_t)B_DIM * OUT_DIM + gid]         = s * x_scale[gid] * w_scale[0];
        out[(size_t)B_DIM * OUT_DIM + B_DIM + gid] = 0.0f;
    }
}

extern "C" void kernel_launch(void* const* d_in, const int* in_sizes, int n_in,
                              void* d_out, int out_size, void* d_ws, size_t ws_size,
                              hipStream_t stream) {
    const float* x_q     = (const float*)d_in[0];
    const float* x_scale = (const float*)d_in[1];
    const float* x_zp    = (const float*)d_in[2];
    const float* w_q     = (const float*)d_in[3];
    const float* w_scale = (const float*)d_in[4];
    const float* fp_bias = (const float*)d_in[5];

    uint8_t* xp     = (uint8_t*)d_ws;
    int8_t*  wp     = (int8_t*)((char*)d_ws + (size_t)B_DIM * IN_DIM);
    int*     rowsum = (int*)((char*)d_ws + (size_t)B_DIM * IN_DIM + (size_t)OUT_DIM * IN_DIM);
    int*     smax   = rowsum + OUT_DIM;

    float* out = (float*)d_out;
    int*   lin = (int*)d_out;

    pack_x_kernel<<<(unsigned)(((size_t)B_DIM * IN_DIM / 4) / 256), 256, 0, stream>>>(
        x_q, (uint32_t*)xp);
    pack_w_kernel<<<OUT_DIM, 256, 0, stream>>>(w_q, wp, rowsum);
    zero_smax_kernel<<<1, 1, 0, stream>>>(smax);

    dim3 g(B_DIM / 128, OUT_DIM / 128);
    gemm_kernel<<<g, 256, 0, stream>>>(xp, wp, rowsum, x_scale, x_zp, fp_bias, w_scale,
                                       lin, smax);

    size_t nq = (size_t)B_DIM * OUT_DIM;
    requant_kernel<<<(unsigned)((nq + 255) / 256), 256, 0, stream>>>(lin, x_scale, w_scale,
                                                                     smax, out);
}
